// LinearAttention_80221399155055
// MI455X (gfx1250) — matrix operand order repaired and checked by reading, not compiled, not run on hardware
//
#include <hip/hip_runtime.h>
#include <hip/hip_bf16.h>

// ---------------------------------------------------------------------------
// LinearAttention for MI455X (gfx1250, wave32, WMMA bf16 16x16x32, f32 accum)
//
// Pipeline (all on `stream`):
//  K0  convert x -> x_bf (transposed [b][n][c]), w_qkv -> bf16, w_out -> bf16
//  K1  QKV GEMM  qkv[b][o][n] = w_qkv[o][c] * x[b][c][n]  (WMMA, async->LDS)
//  K2a softmax over d (q), *SCALE  -> q_bf  [bh][n][d]   (B^T layout for K4)
//  K2b softmax over n (k)          -> k_bf  [bh][d][n]   (A layout for K3)
//  K2c v / n                       -> v_bf  [bh][e][n]   (B^T layout for K3)
//  K3  context^T[e][d] = sum_n k[d][n] v[e][n]            (WMMA)  -> bf16
//  K4  out^T[b][n][h*64+e] = sum_d ctxT[e][d] q[d][n]     (WMMA)  -> bf16
//  K5  y = w_out @ out + b_out, then LayerNorm over C, *g_ln  (WMMA + LDS LN)
// ---------------------------------------------------------------------------

typedef __bf16 bf16_t;
typedef __attribute__((ext_vector_type(16))) __bf16 v16bf;
typedef __attribute__((ext_vector_type(8)))  float  v8f;
typedef int v4i_g __attribute__((vector_size(16)));   // matches builtin param type

#define NB    8
#define CC    256
#define NN    4096
#define HEADS 8
#define DHEAD 64
#define HD    512          // HEADS*DHEAD
#define OQKV  1536         // 3*HD
#define SCALE 0.125f       // 64^-0.5
#define LN_EPS 1e-5f

#define AS1 __attribute__((address_space(1)))
#define AS3 __attribute__((address_space(3)))

// ------------------------- async global->LDS (CDNA5) ------------------------
// GLOBAL_LOAD_ASYNC_TO_LDS_B128: direct memory->LDS DMA, tracked by ASYNCcnt.
// Builtin signature (from hipcc diagnostic): (v4i AS1*, v4i AS3*, imm, imm).

static __device__ __forceinline__ void async_copy16(const bf16_t* g, bf16_t* l) {
#if __has_builtin(__builtin_amdgcn_global_load_async_to_lds_b128)
  __builtin_amdgcn_global_load_async_to_lds_b128(
      (AS1 v4i_g*)g, (AS3 v4i_g*)l, /*imm offset=*/0, /*cpol=*/0);
#else
  const unsigned* s = (const unsigned*)g;
  unsigned* d = (unsigned*)l;
  d[0] = s[0]; d[1] = s[1]; d[2] = s[2]; d[3] = s[3];
#endif
}

static __device__ __forceinline__ void async_wait0() {
#if __has_builtin(__builtin_amdgcn_s_wait_asynccnt)
  __builtin_amdgcn_s_wait_asynccnt(0);
#elif __has_builtin(__builtin_amdgcn_global_load_async_to_lds_b128)
  asm volatile("s_wait_asynccnt 0x0" ::: "memory");
#endif
}

// ------------------------- WMMA helpers ------------------------------------

static __device__ __forceinline__ v8f wmma_bf16(v16bf a, v16bf b, v8f c) {
  // D = A(16x32 bf16) * B(32x16 bf16) + C(16x16 f32)
  return __builtin_amdgcn_wmma_f32_16x16x32_bf16(
      /*neg_a=*/false, a, /*neg_b=*/false, b,
      /*c_mod=*/(short)0, c, /*reuse_a=*/false, /*reuse_b=*/false);
}

// A fragment: 16x32 bf16, source tile row-major [m][k], leading dim ld.
// Layout: lanes 0-15 hold K=0..15 (pairs), lanes 16-31 hold K=16..31 (matches B).
static __device__ __forceinline__ v16bf load_a_frag(const bf16_t* tile, int ld) {
  union { v16bf v; unsigned u[8]; } f;
  unsigned lane = threadIdx.x & 31u;
  unsigned m    = lane & 15u;
  unsigned kb   = (lane >> 4) * 16u;
  const bf16_t* row = tile + (size_t)m * ld;
#pragma unroll
  for (int i = 0; i < 8; ++i)
    f.u[i] = *(const unsigned*)(row + kb + 2u * i);
  return f.v;
}

// B fragment: 32x16 bf16, source is B^T stored row-major [n][k], leading dim ld.
// Layout (contiguous k, same as A): lanes 0-15 hold K=0..15 (pairs), lanes 16-31 hold K=16..31.
static __device__ __forceinline__ v16bf load_b_frag(const bf16_t* tileT, int ld) {
  union { v16bf v; unsigned u[8]; } f;
  unsigned lane = threadIdx.x & 31u;
  unsigned n    = lane & 15u;
  unsigned kb   = (lane >> 4) * 16u;
  const bf16_t* row = tileT + (size_t)n * ld + kb;
#pragma unroll
  for (int i = 0; i < 8; ++i)
    f.u[i] = *(const unsigned*)(row + 2u * i);
  return f.v;
}

// D store (f32, row-major [m][n], leading dim ld).
// ISA C/D layout: lane holds N=lane&15; VGPR i -> M = i + 8*(lane>=16).
static __device__ __forceinline__ void store_d_rowmajor(float* base, int ld, v8f d) {
  unsigned lane = threadIdx.x & 31u;
  unsigned nn   = lane & 15u;
  unsigned mb   = (lane >> 4) * 8u;
#pragma unroll
  for (int i = 0; i < 8; ++i) base[(size_t)(mb + i) * ld + nn] = d[i];
}

// ------------------------- K0: precision conversion ------------------------

__global__ void k_convert(const float* __restrict__ x, const float* __restrict__ wqkv,
                          const float* __restrict__ wout, bf16_t* __restrict__ x_bf,
                          bf16_t* __restrict__ wqkv_bf, bf16_t* __restrict__ wout_bf) {
  size_t idx = (size_t)blockIdx.x * blockDim.x + threadIdx.x;
  if (idx < (size_t)NB * CC * NN) {
    size_t b = idx / ((size_t)CC * NN);
    size_t r = idx % ((size_t)CC * NN);
    size_t c = r / NN, n = r % NN;
    // x [b][c][n] -> x_bf [b][n][c]  (B^T layout for the QKV GEMM)
    x_bf[(b * NN + n) * CC + c] = (bf16_t)x[idx];
  }
  if (idx < (size_t)OQKV * CC) wqkv_bf[idx] = (bf16_t)wqkv[idx];
  if (idx < (size_t)CC * HD)   wout_bf[idx] = (bf16_t)wout[idx];
}

// ------------------------- K1: QKV projection GEMM -------------------------
// Per batch: A = w_qkv (1536x256), B^T = x_bf[b] (4096x256), C = qkv f32.
// Workgroup tile 128(M) x 64(N); 8 waves; wave w -> M rows [16w,16w+16), 4 N tiles.
// Tiles staged with GLOBAL_LOAD_ASYNC_TO_LDS_B128 (ASYNCcnt).

__global__ __launch_bounds__(256) void k_qkv_gemm(const bf16_t* __restrict__ A,
                                                  const bf16_t* __restrict__ Bt,
                                                  float* __restrict__ C) {
  __shared__ bf16_t sA[128][32];   // 8 KB
  __shared__ bf16_t sB[64][32];    // 4 KB
  const int b  = blockIdx.z;
  const int m0 = blockIdx.y * 128;
  const int n0 = blockIdx.x * 64;
  const bf16_t* Bb = Bt + (size_t)b * NN * CC;
  float*        Cb = C + (size_t)b * OQKV * NN;
  const int wave = threadIdx.x >> 5;

  v8f acc[4] = {};
  for (int k0 = 0; k0 < CC; k0 += 32) {
    // Stage A tile 128x32: 512 x 16B chunks, 2 per thread (async -> LDS)
#pragma unroll
    for (int j = 0; j < 2; ++j) {
      int cidx = threadIdx.x * 2 + j;
      int row = cidx >> 2, off = (cidx & 3) * 8;
      async_copy16(A + (size_t)(m0 + row) * CC + k0 + off, &sA[row][off]);
    }
    // Stage B^T tile 64x32: 256 x 16B chunks, 1 per thread (async -> LDS)
    {
      int cidx = threadIdx.x;
      int row = cidx >> 2, off = (cidx & 3) * 8;
      async_copy16(Bb + (size_t)(n0 + row) * CC + k0 + off, &sB[row][off]);
    }
    // Prefetch next K tile toward near caches (global_prefetch_b8)
    if (k0 + 32 < CC) {
      __builtin_prefetch(A + (size_t)(m0 + (threadIdx.x >> 1)) * CC + k0 + 32, 0, 3);
      __builtin_prefetch(Bb + (size_t)(n0 + (threadIdx.x >> 2)) * CC + k0 + 32, 0, 3);
    }
    async_wait0();          // ASYNCcnt == 0: this wave's tiles landed in LDS
    __syncthreads();        // all waves' tiles visible

    v16bf af = load_a_frag(&sA[wave * 16][0], 32);
#pragma unroll
    for (int t = 0; t < 4; ++t) {
      v16bf bf = load_b_frag(&sB[t * 16][0], 32);
      acc[t] = wmma_bf16(af, bf, acc[t]);
    }
    __syncthreads();
  }
#pragma unroll
  for (int t = 0; t < 4; ++t)
    store_d_rowmajor(Cb + (size_t)(m0 + wave * 16) * NN + n0 + t * 16, NN, acc[t]);
}

// ------------------------- K2a: softmax over d (q) -------------------------
// One thread per (bh, n); 64 strided reads, writes q_bf [bh][n][d] (B^T layout).

__global__ void k_softmax_q(const float* __restrict__ qkv, bf16_t* __restrict__ q_bf) {
  int idx = blockIdx.x * blockDim.x + threadIdx.x;      // bh*4096 + n
  int n  = idx & (NN - 1);
  int bh = idx >> 12;
  int b = bh >> 3, h = bh & 7;
  const float* src = qkv + ((size_t)b * OQKV + (size_t)h * DHEAD) * NN + n;
  float vals[DHEAD];
  float m = -3.402823466e38f;
#pragma unroll
  for (int d = 0; d < DHEAD; ++d) {
    vals[d] = src[(size_t)d * NN];
    m = fmaxf(m, vals[d]);
  }
  float s = 0.f;
#pragma unroll
  for (int d = 0; d < DHEAD; ++d) { vals[d] = __expf(vals[d] - m); s += vals[d]; }
  float inv = SCALE / s;
  bf16_t* dst = q_bf + ((size_t)bh * NN + n) * DHEAD;
#pragma unroll
  for (int d = 0; d < DHEAD; ++d) dst[d] = (bf16_t)(vals[d] * inv);
}

// ------------------------- K2b: softmax over n (k) -------------------------
// One workgroup per row (bh,d); 4096-wide reduce; writes k_bf [bh][d][n].

__global__ __launch_bounds__(256) void k_softmax_k(const float* __restrict__ qkv,
                                                   bf16_t* __restrict__ k_bf) {
  __shared__ float red[256];
  const int row = blockIdx.x;              // b*512 + h*64 + d
  const int b   = row >> 9;
  const int hd_ = row & 511;
  const int tid = threadIdx.x;
  const float* src = qkv + ((size_t)b * OQKV + HD + hd_) * NN;

  float m = -3.402823466e38f;
  for (int i = tid; i < NN; i += 256) m = fmaxf(m, src[i]);
  red[tid] = m; __syncthreads();
  for (int s = 128; s > 0; s >>= 1) {
    if (tid < s) red[tid] = fmaxf(red[tid], red[tid + s]);
    __syncthreads();
  }
  m = red[0]; __syncthreads();

  float sum = 0.f;
  for (int i = tid; i < NN; i += 256) sum += __expf(src[i] - m);
  red[tid] = sum; __syncthreads();
  for (int s = 128; s > 0; s >>= 1) {
    if (tid < s) red[tid] += red[tid + s];
    __syncthreads();
  }
  float inv = 1.f / red[0];

  bf16_t* dst = k_bf + (size_t)row * NN;
  for (int i = tid; i < NN; i += 256) dst[i] = (bf16_t)(__expf(src[i] - m) * inv);
}

// ------------------------- K2c: scale v ------------------------------------
// v_bf [bh][e][n] = qkv_v / n   (B^T layout for K3, since B^T[e][n] = v[e][n])

__global__ void k_scale_v(const float* __restrict__ qkv, bf16_t* __restrict__ v_bf) {
  size_t idx = (size_t)blockIdx.x * blockDim.x + threadIdx.x;  // [bh][e][n]
  size_t n  = idx & (NN - 1);
  size_t e  = (idx >> 12) & 63;
  size_t bh = idx >> 18;
  size_t b = bh >> 3, h = bh & 7;
  float v = qkv[((size_t)b * OQKV + 2 * HD + h * DHEAD + e) * NN + n];
  v_bf[idx] = (bf16_t)(v * (1.0f / NN));
}

// ------------------------- K3: context GEMM --------------------------------
// Per (b,h): ctxT[e][d] = sum_n k[d][n] v[e][n].  M=d=64, N=e=64, K=n=4096.

__global__ __launch_bounds__(256) void k_context(const bf16_t* __restrict__ k_bf,
                                                 const bf16_t* __restrict__ v_bf,
                                                 bf16_t* __restrict__ ctxT) {
  const int bh = blockIdx.x;
  const bf16_t* A  = k_bf + (size_t)bh * DHEAD * NN;   // [d][n], ld=NN
  const bf16_t* Bt = v_bf + (size_t)bh * DHEAD * NN;   // B^T=[e][n], ld=NN
  const int wave = threadIdx.x >> 5;
  const int mt = wave & 3;
  const int nb = (wave >> 2) * 2;

  v8f acc0 = {}, acc1 = {};
  for (int k0 = 0; k0 < NN; k0 += 32) {
    v16bf a  = load_a_frag(A + (size_t)mt * 16 * NN + k0, NN);
    v16bf b0 = load_b_frag(Bt + (size_t)(nb + 0) * 16 * NN + k0, NN);
    v16bf b1 = load_b_frag(Bt + (size_t)(nb + 1) * 16 * NN + k0, NN);
    acc0 = wmma_bf16(a, b0, acc0);
    acc1 = wmma_bf16(a, b1, acc1);
  }
  // Transposed bf16 store: D[m=d][n=e] -> ctxT[e*64 + d] (contiguous per lane)
  bf16_t* C = ctxT + (size_t)bh * DHEAD * DHEAD;
  unsigned lane = threadIdx.x & 31u;
  unsigned nn2  = lane & 15u;
  unsigned mb   = (lane >> 4) * 8u;
  bf16_t* p0 = C + (size_t)((nb + 0) * 16 + nn2) * DHEAD + mt * 16 + mb;
  bf16_t* p1 = C + (size_t)((nb + 1) * 16 + nn2) * DHEAD + mt * 16 + mb;
#pragma unroll
  for (int i = 0; i < 8; ++i) { p0[i] = (bf16_t)acc0[i]; p1[i] = (bf16_t)acc1[i]; }
}

// ------------------------- K4: attention output GEMM -----------------------
// Per (b,h): out[e][n] = sum_d ctxT[e][d] q[d][n]. Stored transposed:
// out_bf [b][n][h*64+e]  (B^T layout for K5).

__global__ __launch_bounds__(256) void k_out_gemm(const bf16_t* __restrict__ ctxT,
                                                  const bf16_t* __restrict__ q_bf,
                                                  bf16_t* __restrict__ out_bf) {
  const int bh = blockIdx.y;
  const int b = bh >> 3, h = bh & 7;
  const int n0 = blockIdx.x * 128;
  const bf16_t* A  = ctxT + (size_t)bh * DHEAD * DHEAD;       // [e][d], ld=64
  const bf16_t* Bt = q_bf + ((size_t)bh * NN + n0) * DHEAD;   // [n][d], ld=64
  bf16_t* outb = out_bf + ((size_t)b * NN + n0) * HD + h * DHEAD;
  const int wave = threadIdx.x >> 5;
  const int mt = wave & 3;
  const int nb = (wave >> 2) * 4;

  v8f acc[4] = {};
#pragma unroll
  for (int k0 = 0; k0 < DHEAD; k0 += 32) {
    v16bf a = load_a_frag(A + (size_t)mt * 16 * DHEAD + k0, DHEAD);
#pragma unroll
    for (int t = 0; t < 4; ++t) {
      v16bf bf = load_b_frag(Bt + (size_t)(nb + t) * 16 * DHEAD + k0, DHEAD);
      acc[t] = wmma_bf16(a, bf, acc[t]);
    }
  }
  unsigned lane = threadIdx.x & 31u;
  unsigned nn2  = lane & 15u;
  unsigned mb   = (lane >> 4) * 8u;
#pragma unroll
  for (int t = 0; t < 4; ++t) {
    bf16_t* p = outb + (size_t)((nb + t) * 16 + nn2) * HD + mt * 16 + mb;
#pragma unroll
    for (int i = 0; i < 8; ++i) p[i] = (bf16_t)acc[t][i];
  }
}

// ------------------------- K5: projection + bias + LayerNorm ---------------
// Per (b, 32-column strip): y[0..255][n0..n0+31] via WMMA (K=512), bias add,
// then LayerNorm over the 256 channels entirely in LDS.

__global__ __launch_bounds__(256) void k_proj_ln(const bf16_t* __restrict__ Aw,
                                                 const bf16_t* __restrict__ Bt,
                                                 const float* __restrict__ b_out,
                                                 const float* __restrict__ g_ln,
                                                 float* __restrict__ y) {
  __shared__ float sy[CC * 32];        // 32 KB: full channel strip
  __shared__ float ps[8 * 32], pq[8 * 32];
  __shared__ float s_mean[32], s_inv[32];

  const int b  = blockIdx.y;
  const int n0 = blockIdx.x * 32;
  const bf16_t* Bb = Bt + ((size_t)b * NN + n0) * HD;   // [n][o'], ld=512
  const int wave = threadIdx.x >> 5;

  v8f acc[2][2] = {};
  for (int k0 = 0; k0 < HD; k0 += 32) {
    v16bf a0 = load_a_frag(Aw + (size_t)(wave * 32 + 0)  * HD + k0, HD);
    v16bf a1 = load_a_frag(Aw + (size_t)(wave * 32 + 16) * HD + k0, HD);
    v16bf b0 = load_b_frag(Bb + (size_t)0  * HD + k0, HD);
    v16bf b1 = load_b_frag(Bb + (size_t)16 * HD + k0, HD);
    acc[0][0] = wmma_bf16(a0, b0, acc[0][0]);
    acc[0][1] = wmma_bf16(a0, b1, acc[0][1]);
    acc[1][0] = wmma_bf16(a1, b0, acc[1][0]);
    acc[1][1] = wmma_bf16(a1, b1, acc[1][1]);
  }

  // Store accumulators + bias into LDS strip (row-major [c][j], ld=32)
  {
    unsigned lane = threadIdx.x & 31u;
    unsigned nn2  = lane & 15u;
    unsigned mb   = (lane >> 4) * 8u;
#pragma unroll
    for (int m2 = 0; m2 < 2; ++m2) {
#pragma unroll
      for (int nt = 0; nt < 2; ++nt) {
        int mbase = wave * 32 + m2 * 16;
        float* base = sy + (size_t)mbase * 32 + nt * 16;
#pragma unroll
        for (int i = 0; i < 8; ++i)
          base[(size_t)(mb + i) * 32 + nn2] = acc[m2][nt][i] + b_out[mbase + mb + i];
      }
    }
  }
  __syncthreads();

  // Partial mean/var: thread t -> column t&31, channel block (t>>5)*32..+31
  {
    int col = threadIdx.x & 31;
    int qr  = threadIdx.x >> 5;
    float s = 0.f, ss = 0.f;
#pragma unroll 4
    for (int r = 0; r < 32; ++r) {
      float v = sy[(size_t)(qr * 32 + r) * 32 + col];
      s += v; ss += v * v;
    }
    ps[qr * 32 + col] = s;
    pq[qr * 32 + col] = ss;
  }
  __syncthreads();
  if (threadIdx.x < 32) {
    float S = 0.f, SS = 0.f;
#pragma unroll
    for (int q = 0; q < 8; ++q) { S += ps[q * 32 + threadIdx.x]; SS += pq[q * 32 + threadIdx.x]; }
    float mean = S * (1.f / CC);
    float var  = SS * (1.f / CC) - mean * mean;
    s_mean[threadIdx.x] = mean;
    s_inv[threadIdx.x]  = rsqrtf(var + LN_EPS);
  }
  __syncthreads();

  float* yb = y + (size_t)b * CC * NN + n0;
  for (int idx = threadIdx.x; idx < CC * 32; idx += 256) {
    int c = idx >> 5, j = idx & 31;
    yb[(size_t)c * NN + j] = (sy[idx] - s_mean[j]) * s_inv[j] * g_ln[c];
  }
}

// ------------------------- launcher ----------------------------------------

extern "C" void kernel_launch(void* const* d_in, const int* in_sizes, int n_in,
                              void* d_out, int out_size, void* d_ws, size_t ws_size,
                              hipStream_t stream) {
  const float* x     = (const float*)d_in[0];
  const float* wqkv  = (const float*)d_in[1];
  const float* wout  = (const float*)d_in[2];
  const float* b_out = (const float*)d_in[3];
  const float* g_ln  = (const float*)d_in[4];
  float* y = (float*)d_out;

  // Workspace carve-up (all 256B aligned by construction)
  char* w = (char*)d_ws;
  float* qkv = (float*)w;           w += (size_t)NB * OQKV * NN * 4;      // 201.3 MB
  bf16_t* x_bf = (bf16_t*)w;        w += (size_t)NB * NN * CC * 2;        // 16.8 MB
  bf16_t* wqkv_bf = (bf16_t*)w;     w += (size_t)OQKV * CC * 2;           // 0.79 MB
  bf16_t* wout_bf = (bf16_t*)w;     w += (size_t)CC * HD * 2;             // 0.26 MB
  bf16_t* q_bf = (bf16_t*)w;        w += (size_t)NB * HEADS * NN * DHEAD * 2;  // 33.6 MB
  bf16_t* k_bf = (bf16_t*)w;        w += (size_t)NB * HEADS * DHEAD * NN * 2;  // 33.6 MB
  bf16_t* v_bf = (bf16_t*)w;        w += (size_t)NB * HEADS * DHEAD * NN * 2;  // 33.6 MB
  bf16_t* ctxT = (bf16_t*)w;        w += (size_t)NB * HEADS * DHEAD * DHEAD * 2; // 0.5 MB
  bf16_t* out_bf = (bf16_t*)w;      w += (size_t)NB * NN * HD * 2;        // 33.6 MB
  (void)ws_size; (void)in_sizes; (void)n_in; (void)out_size;

  // K0: conversions (covers x: 8.38M elems; weights handled by low indices)
  k_convert<<<dim3(32768), 256, 0, stream>>>(x, wqkv, wout, x_bf, wqkv_bf, wout_bf);
  // K1: QKV GEMM
  k_qkv_gemm<<<dim3(NN / 64, OQKV / 128, NB), 256, 0, stream>>>(wqkv_bf, x_bf, qkv);
  // K2: softmaxes + v scale
  k_softmax_q<<<dim3((NB * HEADS * NN) / 256), 256, 0, stream>>>(qkv, q_bf);
  k_softmax_k<<<dim3(NB * HEADS * DHEAD), 256, 0, stream>>>(qkv, k_bf);
  k_scale_v<<<dim3((NB * HEADS * DHEAD * NN) / 256), 256, 0, stream>>>(qkv, v_bf);
  // K3: context GEMM (per b,h)
  k_context<<<dim3(NB * HEADS), 256, 0, stream>>>(k_bf, v_bf, ctxT);
  // K4: attention output GEMM
  k_out_gemm<<<dim3(NN / 128, NB * HEADS), 256, 0, stream>>>(ctxT, q_bf, out_bf);
  // K5: projection + bias + LayerNorm
  k_proj_ln<<<dim3(NN / 32, NB), 256, 0, stream>>>(wout_bf, out_bf, b_out, g_ln, y);
}
